// ZhangCapSAREncoder_64141041599139
// MI455X (gfx1250) — matrix operand order repaired and checked by reading, not compiled, not run on hardware
//
#include <hip/hip_runtime.h>
#include <hip/hip_bf16.h>
#include <stdint.h>

typedef _Float16 half8 __attribute__((ext_vector_type(8)));
typedef _Float16 v16h  __attribute__((ext_vector_type(16)));
typedef float    v8f   __attribute__((ext_vector_type(8)));

static __device__ __forceinline__ half8 h8zero() {
    half8 z;
#pragma unroll
    for (int i = 0; i < 8; ++i) z[i] = (_Float16)0.0f;
    return z;
}

static __device__ __forceinline__ half8 ld8(const _Float16* p) {
    return *(const half8*)p;
}

static __device__ __forceinline__ v16h make_v16(half8 lo, half8 hi) {
    v16h r;
#pragma unroll
    for (int i = 0; i < 8; ++i) { r[i] = lo[i]; r[i + 8] = hi[i]; }
    return r;
}

// -------------------- weight format conversion --------------------

// w2 [32][16][9][9] f32 -> w2t [81][32][16] f16
__global__ __launch_bounds__(256) void k_cvt_w2(const float* __restrict__ w2,
                                                _Float16* __restrict__ w2t) {
    int idx = blockIdx.x * 256 + threadIdx.x;   // 41472
    if (idx >= 81 * 32 * 16) return;
    int tap = idx / 512;
    int rem = idx % 512;
    int n = rem / 16, ci = rem % 16;
    w2t[idx] = (_Float16)w2[(n * 16 + ci) * 81 + tap];
}

// wp [64][32][9][9] f32 -> wpt [81][64][32] f16
__global__ __launch_bounds__(256) void k_cvt_wp(const float* __restrict__ wp,
                                                _Float16* __restrict__ wpt) {
    int idx = blockIdx.x * 256 + threadIdx.x;   // 165888
    if (idx >= 81 * 64 * 32) return;
    int tap = idx / 2048;
    int rem = idx % 2048;
    int n = rem / 32, ci = rem % 32;
    wpt[idx] = (_Float16)wp[(n * 32 + ci) * 81 + tap];
}

// w_route f32 [400][10][16][64] -> f16 same layout
__global__ __launch_bounds__(256) void k_cvt_wr(const float* __restrict__ wr,
                                                _Float16* __restrict__ wr16, int n) {
    int idx = blockIdx.x * 256 + threadIdx.x;
    if (idx < n) wr16[idx] = (_Float16)wr[idx];
}

// -------------------- conv1: 1->16, 9x9 valid, direct --------------------
// x [256][1][64][64] f32 -> h1 [256][56][56][16] f16 (NHWC), ReLU
__global__ __launch_bounds__(256) void k_conv1(const float* __restrict__ x,
                                               const float* __restrict__ w1,
                                               const float* __restrict__ b1,
                                               _Float16* __restrict__ h1) {
    __shared__ float w1s[16 * 81];
    int t = threadIdx.x;
    for (int i = t; i < 16 * 81; i += 256) w1s[i] = w1[i];   // [c][tap]
    __syncthreads();

    int gid = blockIdx.x * 256 + t;       // 256*3136 = 802816 exactly
    int b = gid / 3136;
    int pix = gid % 3136;
    int oy = pix / 56, ox = pix % 56;
    const float* xb = x + (size_t)b * 4096;

    float acc[16];
#pragma unroll
    for (int c = 0; c < 16; ++c) acc[c] = b1[c];

    for (int dy = 0; dy < 9; ++dy) {
        for (int dx = 0; dx < 9; ++dx) {
            float xv = xb[(oy + dy) * 64 + (ox + dx)];
            int tap = dy * 9 + dx;
#pragma unroll
            for (int c = 0; c < 16; ++c) acc[c] += xv * w1s[c * 81 + tap];
        }
    }
    _Float16* o = h1 + (size_t)gid * 16;
    half8 lo, hi;
#pragma unroll
    for (int c = 0; c < 8; ++c) {
        float a0 = acc[c];     lo[c] = (_Float16)(a0 > 0.f ? a0 : 0.f);
        float a1 = acc[c + 8]; hi[c] = (_Float16)(a1 > 0.f ? a1 : 0.f);
    }
    *(half8*)o = lo;
    *(half8*)(o + 8) = hi;
}

// -------------------- conv2: 16->32, 9x9 valid, WMMA implicit GEMM ------
// h1 [b][56][56][16] f16 -> h2 [b][48][48][32] f16, ReLU
// K = 16ch * 81taps -> 40 full K=32 chunks (taps 2t, 2t+1) + tail tap 80.
__global__ __launch_bounds__(128) void k_conv2(const _Float16* __restrict__ h1,
                                               const _Float16* __restrict__ w2t,
                                               const float* __restrict__ b2,
                                               _Float16* __restrict__ h2) {
    int wid  = (blockIdx.x * 128 + threadIdx.x) >> 5;   // 36864 waves
    int lane = threadIdx.x & 31;
    int mtile = wid % 144;
    int b     = wid / 144;

    int n_lo = lane & 15;
    int hi   = lane >> 4;
    int kb   = hi * 8;
    int m0   = mtile * 16;
    int mg   = m0 + n_lo;                 // A row = lane&15
    int oy = mg / 48, ox = mg % 48;

    // per-lane base pointers; all tap displacements are compile-time imms
    const _Float16* aptr = h1 + (size_t)b * 56 * 56 * 16 + (oy * 56 + ox) * 16 + kb;
    const _Float16* bptr = w2t + n_lo * 16 + kb;   // tap stride = 512 halves

    v8f acc0, acc1;
    float bias0 = b2[n_lo], bias1 = b2[16 + n_lo];
#pragma unroll
    for (int r = 0; r < 8; ++r) { acc0[r] = bias0; acc1[r] = bias1; }

#pragma unroll
    for (int tch = 0; tch < 40; ++tch) {
        const int tap0 = 2 * tch, tap1 = tap0 + 1;
        const int aoff0 = ((tap0 / 9) * 56 + (tap0 % 9)) * 16;
        const int aoff1 = ((tap1 / 9) * 56 + (tap1 % 9)) * 16;
        const int boff  = tch * 1024;                    // (2 taps) * 512 halves
        v16h A  = make_v16(ld8(aptr + aoff0), ld8(aptr + aoff1));
        v16h B0 = make_v16(ld8(bptr + boff),       ld8(bptr + boff + 512));
        v16h B1 = make_v16(ld8(bptr + boff + 256), ld8(bptr + boff + 768));
        acc0 = __builtin_amdgcn_wmma_f32_16x16x32_f16(false, A, false, B0, (short)0, acc0, false, false);
        acc1 = __builtin_amdgcn_wmma_f32_16x16x32_f16(false, A, false, B1, (short)0, acc1, false, false);
    }
    {   // tail: tap 80 in K[0,16), K[16,32) zero-padded
        const int aoff = ((80 / 9) * 56 + (80 % 9)) * 16;
        v16h A = make_v16(ld8(aptr + aoff), h8zero());
        half8 b0l, b0h, b1l, b1h;
        {   // both lane halves: elements 0-7 = tap 80 ch 8h+s; elements 8-15 = pad
            b0l = ld8(bptr + 40960);      b0h = h8zero();
            b1l = ld8(bptr + 40960 + 256); b1h = h8zero();
        }
        v16h B0 = make_v16(b0l, b0h);
        v16h B1 = make_v16(b1l, b1h);
        acc0 = __builtin_amdgcn_wmma_f32_16x16x32_f16(false, A, false, B0, (short)0, acc0, false, false);
        acc1 = __builtin_amdgcn_wmma_f32_16x16x32_f16(false, A, false, B1, (short)0, acc1, false, false);
    }

    _Float16* h2b = h2 + (size_t)b * 48 * 48 * 32;
#pragma unroll
    for (int r = 0; r < 8; ++r) {
        int m = r + hi * 8;
        int mgr = m0 + m;
        int oyr = mgr / 48, oxr = mgr % 48;
        float v0 = acc0[r]; v0 = v0 > 0.f ? v0 : 0.f;
        float v1 = acc1[r]; v1 = v1 > 0.f ? v1 : 0.f;
        h2b[(oyr * 48 + oxr) * 32 + n_lo]      = (_Float16)v0;
        h2b[(oyr * 48 + oxr) * 32 + 16 + n_lo] = (_Float16)v1;
    }
}

// -------------------- convp: 32->64, 9x9 stride 2, WMMA ------------------
// h2 [b][48][48][32] f16 -> u_pre [b][400][64] f32; one tap == one K=32 chunk
__global__ __launch_bounds__(128) void k_convp(const _Float16* __restrict__ h2,
                                               const _Float16* __restrict__ wpt,
                                               const float* __restrict__ bp,
                                               float* __restrict__ u_pre) {
    int wid  = (blockIdx.x * 128 + threadIdx.x) >> 5;   // 6400 waves
    int lane = threadIdx.x & 31;
    int mtile = wid % 25;
    int b     = wid / 25;

    int n_lo = lane & 15;
    int hi   = lane >> 4;
    int kb   = hi * 8;
    int k0   = hi * 16;
    int m0   = mtile * 16;
    int mg   = m0 + n_lo;
    int oy = mg / 20, ox = mg % 20;

    const _Float16* aptr = h2 + (size_t)b * 48 * 48 * 32 + (oy * 2 * 48 + ox * 2) * 32 + kb;
    const _Float16* bptr = wpt + n_lo * 32 + kb;        // tap stride = 2048 halves

    v8f acc[4];
#pragma unroll
    for (int nt = 0; nt < 4; ++nt) {
        float bias = bp[nt * 16 + n_lo];
#pragma unroll
        for (int r = 0; r < 8; ++r) acc[nt][r] = bias;
    }

#pragma unroll
    for (int tap = 0; tap < 81; ++tap) {
        const int aoff = ((tap / 9) * 48 + (tap % 9)) * 32;
        const int boff = tap * 2048;
        v16h A = make_v16(ld8(aptr + aoff), ld8(aptr + aoff + 16));
#pragma unroll
        for (int nt = 0; nt < 4; ++nt) {
            v16h B = make_v16(ld8(bptr + boff + nt * 512),
                              ld8(bptr + boff + nt * 512 + 16));
            acc[nt] = __builtin_amdgcn_wmma_f32_16x16x32_f16(false, A, false, B, (short)0, acc[nt], false, false);
        }
    }

    float* ub = u_pre + (size_t)b * 400 * 64;
#pragma unroll
    for (int nt = 0; nt < 4; ++nt) {
#pragma unroll
        for (int r = 0; r < 8; ++r) {
            int m = r + hi * 8;
            ub[(size_t)(m0 + m) * 64 + nt * 16 + n_lo] = acc[nt][r];
        }
    }
}

// -------------------- squash: u_pre f32 -> u_s f16 -----------------------
__global__ __launch_bounds__(256) void k_squash(const float* __restrict__ u_pre,
                                                _Float16* __restrict__ u_s) {
    int idx = blockIdx.x * 256 + threadIdx.x;   // 102400 capsules
    const float4* up = (const float4*)(u_pre + (size_t)idx * 64);
    float f[64];
    float n2 = 0.f;
#pragma unroll
    for (int i = 0; i < 16; ++i) {
        float4 v = up[i];
        f[4 * i]     = v.x; f[4 * i + 1] = v.y;
        f[4 * i + 2] = v.z; f[4 * i + 3] = v.w;
        n2 += v.x * v.x + v.y * v.y + v.z * v.z + v.w * v.w;
    }
    float scale = (n2 / (1.f + n2)) / sqrtf(n2 + 1e-8f);
    _Float16* o = u_s + (size_t)idx * 64;
#pragma unroll
    for (int j = 0; j < 8; ++j) {
        half8 h;
#pragma unroll
        for (int k = 0; k < 8; ++k) h[k] = (_Float16)(scale * f[j * 8 + k]);
        ((half8*)o)[j] = h;
    }
}

// -------------------- votes: u_hat[b][p][c][o] f16 via WMMA --------------
// per p: [256(b) x 64(i)] x [64(i) x 160(c,o)]
__global__ __launch_bounds__(128) void k_votes(const _Float16* __restrict__ u_s,
                                               const _Float16* __restrict__ wr16,
                                               _Float16* __restrict__ uhat) {
    int wid  = (blockIdx.x * 128 + threadIdx.x) >> 5;   // 6400 waves
    int lane = threadIdx.x & 31;
    int btile = wid % 16;
    int p     = wid / 16;

    int n_lo = lane & 15;
    int hi   = lane >> 4;
    int kb   = hi * 8;
    int k0   = hi * 16;

    int bm = btile * 16 + n_lo;           // A row = batch element
    const _Float16* ua = u_s + ((size_t)bm * 400 + p) * 64;
    v16h A0 = make_v16(ld8(ua + k0),      ld8(ua + k0 + 8));       // i in [0,32)
    v16h A1 = make_v16(ld8(ua + 32 + k0), ld8(ua + 32 + k0 + 8));  // i in [32,64)

    const _Float16* wb0 = wr16 + (((size_t)p * 10) * 16 + n_lo) * 64 + k0;  // +c*1024
    _Float16* ub = uhat + ((size_t)(btile * 16) * 400 + p) * 160 + n_lo;    // row stride 400*160

#pragma unroll
    for (int c = 0; c < 10; ++c) {
        v8f acc;
#pragma unroll
        for (int r = 0; r < 8; ++r) acc[r] = 0.f;
        const _Float16* wb = wb0 + c * 1024;
        v16h B0 = make_v16(ld8(wb),      ld8(wb + 8));
        v16h B1 = make_v16(ld8(wb + 32), ld8(wb + 40));
        acc = __builtin_amdgcn_wmma_f32_16x16x32_f16(false, A0, false, B0, (short)0, acc, false, false);
        acc = __builtin_amdgcn_wmma_f32_16x16x32_f16(false, A1, false, B1, (short)0, acc, false, false);
#pragma unroll
        for (int r = 0; r < 8; ++r) {
            size_t row = (size_t)(r + hi * 8) * 400 * 160;   // batch rows within tile
            ub[row + c * 16] = (_Float16)acc[r];
        }
    }
}

// -------------------- routing: one block per batch element ---------------
static __device__ __forceinline__ void async_ld_b128(uint32_t lds_off, uint32_t goff,
                                                     const void* base) {
    asm volatile("global_load_async_to_lds_b128 %0, %1, %2"
                 :: "v"(lds_off), "v"(goff), "s"(base) : "memory");
}

__global__ __launch_bounds__(256) void k_route(const _Float16* __restrict__ uhat,
                                               float* __restrict__ out) {
    extern __shared__ char smem[];
    _Float16* uh  = (_Float16*)smem;                 // [400*160] f16 = 128000 B
    float*    blg = (float*)(smem + 128000);         // [400*10]
    float*    cc  = (float*)(smem + 144000);         // [400*10]
    float*    sv  = (float*)(smem + 160000);         // [160]
    float*    vv  = (float*)(smem + 160640);         // [160]

    int b = blockIdx.x;
    int t = threadIdx.x;
    const _Float16* src = uhat + (size_t)b * 64000;

    // async-stage u_hat[b] (128 KB) into LDS
    uint32_t lds0 = (uint32_t)(uintptr_t)uh;
    for (int i = t; i < 8000; i += 256)
        async_ld_b128(lds0 + (uint32_t)i * 16u, (uint32_t)i * 16u, src);
    asm volatile("s_wait_asynccnt 0" ::: "memory");

    for (int i = t; i < 4000; i += 256) blg[i] = 0.f;
    __syncthreads();

    for (int it = 0; it < 3; ++it) {
        // coupling coefficients: softmax over the 10 classes, per p
        for (int p = t; p < 400; p += 256) {
            float bl[10], e[10];
            float mx = -1e30f;
#pragma unroll
            for (int c = 0; c < 10; ++c) { bl[c] = blg[p * 10 + c]; mx = fmaxf(mx, bl[c]); }
            float s = 0.f;
#pragma unroll
            for (int c = 0; c < 10; ++c) { e[c] = __expf(bl[c] - mx); s += e[c]; }
            float inv = 1.f / s;
#pragma unroll
            for (int c = 0; c < 10; ++c) cc[p * 10 + c] = e[c] * inv;
        }
        __syncthreads();

        // s[c,o] = sum_p cc[p,c] * u_hat[p,c,o]   (thread t owns n = c*16+o)
        if (t < 160) {
            int c = t >> 4;
            float acc = 0.f;
            for (int p = 0; p < 400; ++p)
                acc += cc[p * 10 + c] * (float)uh[p * 160 + t];
            sv[t] = acc;
        }
        __syncthreads();

        // v = squash(s)
        if (t < 10) {
            float n2 = 0.f;
#pragma unroll
            for (int o = 0; o < 16; ++o) { float x = sv[t * 16 + o]; n2 += x * x; }
            float scale = (n2 / (1.f + n2)) / sqrtf(n2 + 1e-8f);
#pragma unroll
            for (int o = 0; o < 16; ++o) vv[t * 16 + o] = scale * sv[t * 16 + o];
        }
        __syncthreads();

        // b_logits += <u_hat, v>
        for (int idx = t; idx < 4000; idx += 256) {
            int p = idx / 10, c = idx % 10;
            float d = 0.f;
#pragma unroll
            for (int o = 0; o < 16; ++o)
                d += (float)uh[p * 160 + c * 16 + o] * vv[c * 16 + o];
            blg[idx] += d;
        }
        __syncthreads();
    }

    if (t < 10) {
        float n2 = 0.f;
#pragma unroll
        for (int o = 0; o < 16; ++o) { float x = vv[t * 16 + o]; n2 += x * x; }
        out[b * 10 + t] = sqrtf(n2 + 1e-8f);
    }
}

// -------------------- host launcher --------------------------------------

extern "C" void kernel_launch(void* const* d_in, const int* in_sizes, int n_in,
                              void* d_out, int out_size, void* d_ws, size_t ws_size,
                              hipStream_t stream) {
    (void)in_sizes; (void)n_in; (void)out_size; (void)ws_size;
    const float* x  = (const float*)d_in[0];
    const float* w1 = (const float*)d_in[1];
    const float* b1 = (const float*)d_in[2];
    const float* w2 = (const float*)d_in[3];
    const float* b2 = (const float*)d_in[4];
    const float* wp = (const float*)d_in[5];
    const float* bp = (const float*)d_in[6];
    const float* wr = (const float*)d_in[7];
    float* out = (float*)d_out;
    char* ws = (char*)d_ws;

    _Float16* h1    = (_Float16*)(ws + 0);            // 256*56*56*16 f16 = 25,690,112
    _Float16* h2    = (_Float16*)(ws + 25690112);     // 256*48*48*32 f16 = 37,748,736
    float*    u_pre = (float*)   (ws + 63438848);     // 256*400*64  f32 = 26,214,400
    _Float16* u_s   = (_Float16*)(ws + 89653248);     // 256*400*64  f16 = 13,107,200
    _Float16* w2t   = (_Float16*)(ws + 102760448);    // 81*32*16    f16 = 82,944
    _Float16* wpt   = (_Float16*)(ws + 102843392);    // 81*64*32    f16 = 331,776
    _Float16* wr16  = (_Float16*)(ws + 103175168);    // 400*10*16*64 f16 = 8,192,000
    _Float16* uhat  = (_Float16*)(ws + 111367168);    // 256*400*160 f16 = 32,768,000

    k_cvt_w2<<<(81 * 32 * 16 + 255) / 256, 256, 0, stream>>>(w2, w2t);
    k_cvt_wp<<<(81 * 64 * 32 + 255) / 256, 256, 0, stream>>>(wp, wpt);
    k_cvt_wr<<<(4096000 + 255) / 256, 256, 0, stream>>>(wr, wr16, 4096000);

    k_conv1<<<3136, 256, 0, stream>>>(x, w1, b1, h1);          // 256*56*56 pixels
    k_conv2<<<9216, 128, 0, stream>>>(h1, w2t, b2, h2);        // 36864 waves
    k_convp<<<1600, 128, 0, stream>>>(h2, wpt, bp, u_pre);     // 6400 waves
    k_squash<<<400, 256, 0, stream>>>(u_pre, u_s);             // 102400 capsules
    k_votes<<<1600, 128, 0, stream>>>(u_s, wr16, uhat);        // 6400 waves
    k_route<<<256, 256, 161280, stream>>>(uhat, out);          // one WG per image
}
